// BILSTM_SOFTMAX_53549652247294
// MI455X (gfx1250) — matrix operand order repaired and checked by reading, not compiled, not run on hardware
//
#include <hip/hip_runtime.h>
#include <hip/hip_bf16.h>
#include <stdint.h>

// ---------------------------------------------------------------------------
// BiLSTM tagger on gfx1250 (MI455X), f16 WMMA with f32 accumulation.
//
// Shapes: B=64, S=256, L=16, CE=50, H=300, E=100, NCLS=50.
// BS = B*S = 16384 rows for the "flattened sequence" GEMMs.
//
// WMMA tile: v_wmma_f32_16x16x32_f16  (M=16, N=16, K=32).
//   A fragment (16x32 f16): lane = hi*16 + r (r=0..15).
//     halves 0..7  = A[r][kc*32 + hi*16 + 0..7]
//     halves 8..15 = A[r][kc*32 + hi*16 + 8 + 0..7]
//   B fragment (32x16 f16): lane holds column n=r, K-values hi*16 + 0..15.
//     We PREPACK weights so each lane's 16 halves are contiguous (32B load).
//   C/D fragment (16x16 f32): vgpr v, lane = hi*16+r  ->  (row = v + 8*hi, col = r)
// ---------------------------------------------------------------------------

typedef __attribute__((ext_vector_type(16))) _Float16 v16h;
typedef __attribute__((ext_vector_type(8)))  _Float16 v8h;
typedef __attribute__((ext_vector_type(8)))  float    v8f;

#define BB   64
#define SS   256
#define LL   16
#define BSR  16384   // B*S

__device__ __forceinline__ float sigf(float x) { return 1.0f / (1.0f + expf(-x)); }

// ---------------------------------------------------------------------------
// Generic WMMA GEMM:  Out[M x N] = A[M x Kpad](f16) * Bpacked[Kpad x N](f16)
//                                  (+ Cinit)   (+ bias[n])
// grid.x = Ntiles, grid.y = Mtiles/4, block = 128 (4 waves, one M-tile each).
// Bpacked layout: ((kc*Ntiles + ntile)*32 + lane)*16 + j  (fragment-native).
// ---------------------------------------------------------------------------
__global__ void __launch_bounds__(128)
wmma_gemm(const _Float16* __restrict__ A, int lda,
          const _Float16* __restrict__ Bp, int NtilesB,
          const float* __restrict__ Cinit, int ldc,
          const float* __restrict__ bias,
          float* __restrict__ Out, int ldo,
          int Kchunks)
{
    const int ntile = blockIdx.x;
    const int mtile = blockIdx.y * 4 + (threadIdx.x >> 5);
    const int lane  = threadIdx.x & 31;
    const int r     = lane & 15;
    const int hi    = lane >> 4;

    v8f acc;
    if (Cinit) {
        #pragma unroll
        for (int v = 0; v < 8; ++v)
            acc[v] = Cinit[(size_t)(mtile * 16 + v + 8 * hi) * ldc + ntile * 16 + r];
    } else {
        #pragma unroll
        for (int v = 0; v < 8; ++v) acc[v] = 0.0f;
    }
    if (bias) {
        const float bv = bias[ntile * 16 + r];
        #pragma unroll
        for (int v = 0; v < 8; ++v) acc[v] += bv;
    }

    const _Float16* arow = A + (size_t)(mtile * 16 + r) * lda;
    for (int kc = 0; kc < Kchunks; ++kc) {
        const v8h a0 = *(const v8h*)(arow + kc * 32 + hi * 16);
        const v8h a1 = *(const v8h*)(arow + kc * 32 + hi * 16 + 8);
        v16h a;
        #pragma unroll
        for (int q = 0; q < 8; ++q) { a[q] = a0[q]; a[q + 8] = a1[q]; }
        const v16h b = *(const v16h*)(Bp + (((size_t)kc * NtilesB + ntile) * 32 + lane) * 16);
        acc = __builtin_amdgcn_wmma_f32_16x16x32_f16(
                  false, a, false, b, (short)0, acc, false, false);
    }

    #pragma unroll
    for (int v = 0; v < 8; ++v)
        Out[(size_t)(mtile * 16 + v + 8 * hi) * ldo + ntile * 16 + r] = acc[v];
}

// ---------------------------------------------------------------------------
// Weight repack into fragment-native B layout.
// B[k][n] = W1[n*K1 + (k-off1)]   for k in [off1, off1+K1)
//         = W2[n*K2 + (k-off2)]   for k in [off2, off2+K2)   (optional)
//         = 0 otherwise / n >= Nsrc.
// ---------------------------------------------------------------------------
__global__ void pack_B(const float* __restrict__ W1, int K1, int off1,
                       const float* __restrict__ W2, int K2, int off2,
                       int Nsrc, int Kpad, int Npad, _Float16* __restrict__ Bp)
{
    const int idx = blockIdx.x * blockDim.x + threadIdx.x;
    if (idx >= Kpad * Npad) return;
    const int k = idx / Npad;
    const int n = idx % Npad;
    float v = 0.0f;
    if (n < Nsrc) {
        if (k >= off1 && k < off1 + K1)               v = W1[(size_t)n * K1 + (k - off1)];
        else if (W2 && k >= off2 && k < off2 + K2)    v = W2[(size_t)n * K2 + (k - off2)];
    }
    const int kc = k >> 5, kin = k & 31, khi = kin >> 4, j = kin & 15;
    const int ntile = n >> 4, r = n & 15;
    const int Ntiles = Npad >> 4;
    Bp[(((size_t)kc * Ntiles + ntile) * 32 + (khi * 16 + r)) * 16 + j] = (_Float16)v;
}

__global__ void prep_bias(const float* __restrict__ b1, const float* __restrict__ b2,
                          int Nsrc, int Npad, float* __restrict__ out)
{
    const int i = blockIdx.x * blockDim.x + threadIdx.x;
    if (i >= Npad) return;
    float v = 0.0f;
    if (i < Nsrc) { v = b1[i]; if (b2) v += b2[i]; }
    out[i] = v;
}

__global__ void fill_zero(uint32_t* __restrict__ p, size_t n)
{
    size_t i = (size_t)blockIdx.x * blockDim.x + threadIdx.x;
    const size_t stride = (size_t)gridDim.x * blockDim.x;
    for (; i < n; i += stride) p[i] = 0u;
}

// ---------------------------------------------------------------------------
// Embedding gathers.
// ---------------------------------------------------------------------------
__global__ void char_embed_fill(const int* __restrict__ ids,       // [BS][L]
                                const float* __restrict__ emb,     // [CVOCAB][50]
                                _Float16* __restrict__ Xc0)        // [L][BS][128]
{
    const int idx = blockIdx.x * blockDim.x + threadIdx.x;         // BS*L*64
    if (idx >= BSR * LL * 64) return;
    const int j = idx & 63;
    if (j >= 50) return;
    const int rem = idx >> 6;
    const int l  = rem & 15;
    const int bs = rem >> 4;
    const int id = ids[bs * LL + l];
    Xc0[((size_t)l * BSR + bs) * 128 + j] = (_Float16)emb[(size_t)id * 50 + j];
}

__global__ void word_embed_fill(const int* __restrict__ ids,       // [B][S]
                                const float* __restrict__ emb,     // [VOCAB][100]
                                _Float16* __restrict__ seq0)       // [S*B][160]
{
    const int m = blockIdx.x;            // m = s*B + b
    const int j = threadIdx.x;           // block 128
    if (j >= 100) return;
    const int b = m & 63, s = m >> 6;
    const int id = ids[b * SS + s];
    seq0[(size_t)m * 160 + j] = (_Float16)emb[(size_t)id * 100 + j];
}

// ---------------------------------------------------------------------------
// Char LSTM pointwise step (H=50). gates f32 [BS][208] order i,f,g,o.
// Writes h (f16) to: next-step slab h-cols (50..99), next-layer x-cols (0..49),
// and (final step of top layer) scattered into seq0 cols 100..149.
// ---------------------------------------------------------------------------
__global__ void char_lstm_step(const float* __restrict__ gates,
                               float* __restrict__ c,               // [BS][64]
                               _Float16* __restrict__ dst_next,
                               _Float16* __restrict__ dst_layer,
                               _Float16* __restrict__ dst_feat)
{
    const int idx = blockIdx.x * blockDim.x + threadIdx.x;          // BS*64
    const int row = idx >> 6, j = idx & 63;
    if (row >= BSR || j >= 50) return;
    const float* g = gates + (size_t)row * 208;
    const float iv = sigf(g[j]);
    const float fv = sigf(g[50 + j]);
    const float gv = tanhf(g[100 + j]);
    const float ov = sigf(g[150 + j]);
    const float cn = fv * c[(size_t)row * 64 + j] + iv * gv;
    c[(size_t)row * 64 + j] = cn;
    const float h = ov * tanhf(cn);
    const _Float16 hh = (_Float16)h;
    if (dst_next)  dst_next[(size_t)row * 128 + 50 + j] = hh;
    if (dst_layer) dst_layer[(size_t)row * 128 + j]     = hh;
    if (dst_feat) {
        const int b = row >> 8;          // row = b*S + s
        const int s = row & 255;
        dst_feat[((size_t)s * BB + b) * 160 + 100 + j] = hh;
    }
}

// ---------------------------------------------------------------------------
// BiLSTM pointwise step (H=300). gates f32 [64][1200] order i,f,g,o.
// Updates c [64][320], h_state f16 [64][320], and the per-step output slab.
// ---------------------------------------------------------------------------
__global__ void bilstm_step(const float* __restrict__ gates,
                            float* __restrict__ c,
                            _Float16* __restrict__ h_state,
                            _Float16* __restrict__ out_seq)   // pre-offset: slab + coloff
{
    const int row = blockIdx.x;          // 64
    const int j = threadIdx.x;           // block 320
    if (j >= 300) return;
    const float* g = gates + (size_t)row * 1200;
    const float iv = sigf(g[j]);
    const float fv = sigf(g[300 + j]);
    const float gv = tanhf(g[600 + j]);
    const float ov = sigf(g[900 + j]);
    const float cn = fv * c[(size_t)row * 320 + j] + iv * gv;
    c[(size_t)row * 320 + j] = cn;
    const float h = ov * tanhf(cn);
    const _Float16 hh = (_Float16)h;
    h_state[(size_t)row * 320 + j] = hh;
    out_seq[(size_t)row * 608 + j] = hh;
}

// ---------------------------------------------------------------------------
// log_softmax over 50 classes; logits [S*B][64] -> out [B][S][50].
// ---------------------------------------------------------------------------
__global__ void logsoftmax_kernel(const float* __restrict__ logits,
                                  float* __restrict__ out)
{
    const int m = blockIdx.x * blockDim.x + threadIdx.x;
    if (m >= BSR) return;
    const float* row = logits + (size_t)m * 64;
    float mx = row[0];
    for (int cix = 1; cix < 50; ++cix) mx = fmaxf(mx, row[cix]);
    float s = 0.0f;
    for (int cix = 0; cix < 50; ++cix) s += expf(row[cix] - mx);
    const float lse = mx + logf(s);
    const int b = m & 63, sidx = m >> 6;
    float* o = out + ((size_t)b * SS + sidx) * 50;
    for (int cix = 0; cix < 50; ++cix) o[cix] = row[cix] - lse;
}

// ---------------------------------------------------------------------------
// Host orchestration.
// ---------------------------------------------------------------------------
extern "C" void kernel_launch(void* const* d_in, const int* in_sizes, int n_in,
                              void* d_out, int out_size, void* d_ws, size_t ws_size,
                              hipStream_t stream)
{
    (void)in_sizes; (void)n_in; (void)out_size; (void)ws_size;

    const int*   word_ids   = (const int*)d_in[0];
    const int*   char_ids   = (const int*)d_in[1];
    const float* word_embed = (const float*)d_in[2];
    const float* char_embed = (const float*)d_in[3];
    const float* cWih  = (const float*)d_in[4];
    const float* cWhh  = (const float*)d_in[5];
    const float* cbih  = (const float*)d_in[6];
    const float* cbhh  = (const float*)d_in[7];
    const float* b0Wih = (const float*)d_in[8];
    const float* b0Whh = (const float*)d_in[9];
    const float* b0bih = (const float*)d_in[10];
    const float* b0bhh = (const float*)d_in[11];
    const float* b1Wih = (const float*)d_in[12];
    const float* b1Whh = (const float*)d_in[13];
    const float* b1bih = (const float*)d_in[14];
    const float* b1bhh = (const float*)d_in[15];
    const float* fcW   = (const float*)d_in[16];
    const float* fcb   = (const float*)d_in[17];

    // ---- workspace bump allocator ----
    size_t off = 0;
    auto alloc = [&](size_t bytes) -> void* {
        void* p = (char*)d_ws + off;
        off += (bytes + 255) & ~(size_t)255;
        return p;
    };
    _Float16* Xc0    = (_Float16*)alloc((size_t)LL * BSR * 128 * 2);
    _Float16* Xc1    = (_Float16*)alloc((size_t)LL * BSR * 128 * 2);
    float*    gatesC = (float*)   alloc((size_t)BSR * 208 * 4);
    float*    cChar  = (float*)   alloc((size_t)BSR * 64 * 4);
    _Float16* WcP0   = (_Float16*)alloc((size_t)128 * 208 * 2);
    _Float16* WcP1   = (_Float16*)alloc((size_t)128 * 208 * 2);
    float*    biasC0 = (float*)   alloc(208 * 4);
    float*    biasC1 = (float*)   alloc(208 * 4);
    _Float16* seq0   = (_Float16*)alloc((size_t)BSR * 160 * 2);
    _Float16* h0buf  = (_Float16*)alloc((size_t)BSR * 608 * 2);
    _Float16* h1buf  = (_Float16*)alloc((size_t)BSR * 608 * 2);
    float*    xg     = (float*)   alloc((size_t)BSR * 1200 * 4);
    _Float16* WihP   = (_Float16*)alloc((size_t)608 * 1200 * 2);
    _Float16* WhhP   = (_Float16*)alloc((size_t)320 * 1200 * 2);
    float*    biasB  = (float*)   alloc(1200 * 4);
    _Float16* hState = (_Float16*)alloc((size_t)BB * 320 * 2);
    float*    cState = (float*)   alloc((size_t)BB * 320 * 4);
    float*    gatesB = (float*)   alloc((size_t)BB * 1200 * 4);
    _Float16* fcP    = (_Float16*)alloc((size_t)608 * 64 * 2);
    float*    fcbP   = (float*)   alloc(64 * 4);
    float*    logits = (float*)   alloc((size_t)BSR * 64 * 4);

    auto zero32 = [&](void* p, size_t words) {
        fill_zero<<<dim3(2048), 256, 0, stream>>>((uint32_t*)p, words);
    };
    auto gemm = [&](const _Float16* A, int lda, const _Float16* Bp, int Ntiles,
                    const float* Cinit, int ldc, const float* bias,
                    float* Out, int ldo, int Kc, int Mtiles) {
        wmma_gemm<<<dim3(Ntiles, Mtiles / 4), 128, 0, stream>>>(
            A, lda, Bp, Ntiles, Cinit, ldc, bias, Out, ldo, Kc);
    };
    auto pack = [&](const float* W1, int K1, int off1,
                    const float* W2, int K2, int off2,
                    int Nsrc, int Kpad, int Npad, _Float16* Bp) {
        const int total = Kpad * Npad;
        pack_B<<<dim3((total + 255) / 256), 256, 0, stream>>>(
            W1, K1, off1, W2, K2, off2, Nsrc, Kpad, Npad, Bp);
    };

    // ---- zero all buffers whose pad/initial regions are read ----
    zero32(Xc0,   (size_t)LL * BSR * 128 / 2);
    zero32(Xc1,   (size_t)LL * BSR * 128 / 2);
    zero32(seq0,  (size_t)BSR * 160 / 2);
    zero32(h0buf, (size_t)BSR * 608 / 2);
    zero32(h1buf, (size_t)BSR * 608 / 2);

    // ---- embeddings ----
    char_embed_fill<<<dim3(BSR * LL * 64 / 256), 256, 0, stream>>>(char_ids, char_embed, Xc0);
    word_embed_fill<<<dim3(BSR), 128, 0, stream>>>(word_ids, word_embed, seq0);

    // ---- char LSTM combined weights: B rows 0..49 = W_ih^T, 50..99 = W_hh^T ----
    pack(cWih,           50, 0, cWhh,           50, 50, 200, 128, 208, WcP0);
    pack(cWih + 200 * 50, 50, 0, cWhh + 200 * 50, 50, 50, 200, 128, 208, WcP1);
    prep_bias<<<dim3(1), 256, 0, stream>>>(cbih,       cbhh,       200, 208, biasC0);
    prep_bias<<<dim3(1), 256, 0, stream>>>(cbih + 200, cbhh + 200, 200, 208, biasC1);

    // ---- char LSTM: 2 layers x 16 steps, combined [x|h] GEMM per step ----
    for (int l = 0; l < 2; ++l) {
        _Float16* X  = (l == 0) ? Xc0 : Xc1;
        _Float16* Bw = (l == 0) ? WcP0 : WcP1;
        float*    bi = (l == 0) ? biasC0 : biasC1;
        zero32(cChar, (size_t)BSR * 64);
        for (int t = 0; t < LL; ++t) {
            gemm(X + (size_t)t * BSR * 128, 128, Bw, 13,
                 nullptr, 0, bi, gatesC, 208, 4, BSR / 16);
            _Float16* dnext  = (t < LL - 1) ? X + (size_t)(t + 1) * BSR * 128 : nullptr;
            _Float16* dlayer = (l == 0) ? Xc1 + (size_t)t * BSR * 128 : nullptr;
            _Float16* dfeat  = (l == 1 && t == LL - 1) ? seq0 : nullptr;
            char_lstm_step<<<dim3(BSR * 64 / 256), 256, 0, stream>>>(
                gatesC, cChar, dnext, dlayer, dfeat);
        }
    }

    // ---- 2-layer bidirectional LSTM over words ----
    for (int l = 0; l < 2; ++l) {
        const int K    = (l == 0) ? 150 : 600;
        const int Kpad = (l == 0) ? 160 : 608;
        for (int d = 0; d < 2; ++d) {
            const float* Wih = ((l == 0) ? b0Wih : b1Wih) + (size_t)d * 1200 * K;
            const float* Whh = ((l == 0) ? b0Whh : b1Whh) + (size_t)d * 1200 * 300;
            const float* bih = ((l == 0) ? b0bih : b1bih) + d * 1200;
            const float* bhh = ((l == 0) ? b0bhh : b1bhh) + d * 1200;

            pack(Wih, K,   0, nullptr, 0, 0, 1200, Kpad, 1200, WihP);
            pack(Whh, 300, 0, nullptr, 0, 0, 1200, 320,  1200, WhhP);
            prep_bias<<<dim3(5), 256, 0, stream>>>(bih, bhh, 1200, 1200, biasB);

            // xg = x * W_ih^T + b  for all 256 steps at once
            const _Float16* Ain = (l == 0) ? seq0 : h0buf;
            const int lda = (l == 0) ? 160 : 608;
            gemm(Ain, lda, WihP, 75, nullptr, 0, biasB, xg, 1200, Kpad / 32, BSR / 16);

            zero32(hState, (size_t)BB * 320 / 2);
            zero32(cState, (size_t)BB * 320);

            _Float16* outSeq = ((l == 0) ? h0buf : h1buf) + d * 300;
            for (int step = 0; step < SS; ++step) {
                const int t = (d == 0) ? step : (SS - 1 - step);
                gemm(hState, 320, WhhP, 75,
                     xg + (size_t)t * BB * 1200, 1200, nullptr,
                     gatesB, 1200, 10, BB / 16);
                bilstm_step<<<dim3(BB), 320, 0, stream>>>(
                    gatesB, cState, hState, outSeq + (size_t)t * BB * 608);
            }
        }
    }

    // ---- FC + log_softmax ----
    pack(fcW, 600, 0, nullptr, 0, 0, 50, 608, 64, fcP);
    prep_bias<<<dim3(1), 64, 0, stream>>>(fcb, nullptr, 50, 64, fcbP);
    gemm(h1buf, 608, fcP, 4, nullptr, 0, fcbP, logits, 64, 19, BSR / 16);
    logsoftmax_kernel<<<dim3(BSR / 256), 256, 0, stream>>>(logits, (float*)d_out);
}
